// NonLocalAttention_14439680049652
// MI455X (gfx1250) — compile-verified, matrix operand order checked by reading, not run on hardware
//
#include <hip/hip_runtime.h>
#include <stdint.h>

// ---------------------------------------------------------------------------
// NonLocalAttention block for MI455X (gfx1250), wave32 + WMMA bf16 pipeline,
// LDS-staged double-buffered GEMMs with async global->LDS copies.
//   B=4, S=2048, D=1024, H=16, Hd=64
// ---------------------------------------------------------------------------

typedef __attribute__((ext_vector_type(16))) __bf16 v16bf;
typedef __attribute__((ext_vector_type(8)))  __bf16 v8bf;
typedef __attribute__((ext_vector_type(8)))  float  v8f;

static constexpr int Bc   = 4;
static constexpr int Sc   = 2048;
static constexpr int Dc   = 1024;
static constexpr int Hc   = 16;
static constexpr int Hd   = 64;
static constexpr int Mtot = Bc * Sc;      // 8192 rows

// ---------------- CDNA5 async global->LDS (ASYNCcnt) ------------------------
__device__ __forceinline__ void async_b128(const void* g, const void* lds) {
  asm volatile("global_load_async_to_lds_b128 %0, %1, off"
               :: "v"((uint32_t)(uintptr_t)lds), "v"(g)
               : "memory");
}
__device__ __forceinline__ void wait_async0() {
  asm volatile("s_wait_asynccnt 0x0" ::: "memory");
}

// ---------------- fragment helpers -----------------------------------------
__device__ __forceinline__ v8bf pack8(float4 a, float4 b) {
  v8bf o;
  o[0] = (__bf16)a.x; o[1] = (__bf16)a.y; o[2] = (__bf16)a.z; o[3] = (__bf16)a.w;
  o[4] = (__bf16)b.x; o[5] = (__bf16)b.y; o[6] = (__bf16)b.z; o[7] = (__bf16)b.w;
  return o;
}
__device__ __forceinline__ v16bf cat16(v8bf lo, v8bf hi) {
  return __builtin_shufflevector(lo, hi, 0,1,2,3,4,5,6,7,8,9,10,11,12,13,14,15);
}
// 16-bit A/B fragment: lane = {r=lane&15 (row/col), half=lane>>4}.
// Elements 0..7 = K (half*8)+0..7, elements 8..15 = K 16+(half*8)+0..7.
__device__ __forceinline__ v16bf frag_from_bf16(const __bf16* p) {
  v8bf lo = *(const v8bf*)(p);
  v8bf hi = *(const v8bf*)(p + 16);
  return cat16(lo, hi);
}
__device__ __forceinline__ v8f wmma_bf16(v16bf a, v16bf b, v8f c) {
  return __builtin_amdgcn_wmma_f32_16x16x32_bf16(
      false, a, false, b, (short)0, c, false, false);
}
__device__ __forceinline__ v8f zero8() {
  v8f z = {0.f,0.f,0.f,0.f,0.f,0.f,0.f,0.f};
  return z;
}
__device__ __forceinline__ float rmax16(float v) {
  v = fmaxf(v, __shfl_xor(v, 1));
  v = fmaxf(v, __shfl_xor(v, 2));
  v = fmaxf(v, __shfl_xor(v, 4));
  v = fmaxf(v, __shfl_xor(v, 8));
  return v;
}
__device__ __forceinline__ float rsum16(float v) {
  v += __shfl_xor(v, 1);
  v += __shfl_xor(v, 2);
  v += __shfl_xor(v, 4);
  v += __shfl_xor(v, 8);
  return v;
}

// ---------------------------------------------------------------------------
// Kernel 1: fused QKV projection.  out = x @ W.T + b  (NT GEMM).
// WG = 256 thr (8 waves), tile 64(M) x 64(N) for all three weights.
// x tile async-staged fp32 into LDS; W tiles staged once as bf16 (reused 4x).
// Q,K stored bf16 [B,H,S,Hd]; V stored bf16 transposed [B,H,Hd,S].
// ---------------------------------------------------------------------------
__global__ __launch_bounds__(256) void qkv_kernel(
    const float* __restrict__ x,
    const float* __restrict__ Wq, const float* __restrict__ bq,
    const float* __restrict__ Wk, const float* __restrict__ bk,
    const float* __restrict__ Wv, const float* __restrict__ bv,
    __bf16* __restrict__ Qb, __bf16* __restrict__ Kb, __bf16* __restrict__ Vb) {
  __shared__ __align__(16) float  Xs[2][64][36];      // 18.0 KB, padded rows
  __shared__ __align__(16) __bf16 Ws[3][2][64][40];   // 30.0 KB, padded rows

  const int tid  = threadIdx.x;
  const int wave = tid >> 5;
  const int lane = tid & 31;
  const int r    = lane & 15;
  const int half = lane >> 4;
  const int mw   = wave & 3;        // M sub-tile (16 rows)
  const int nw   = wave >> 2;       // N sub-tile (32 cols)
  const int m0   = blockIdx.y * 64;
  const int n0   = blockIdx.x * 64;

  const float* Wp[3] = {Wq, Wk, Wv};

  // ---- staging ----
  auto stageX = [&](int buf, int k0) {
#pragma unroll
    for (int i = 0; i < 2; ++i) {
      int c   = tid + i * 256;      // 0..511 chunks of 16B
      int row = c >> 3;
      int co  = (c & 7) * 4;
      async_b128(x + (size_t)(m0 + row) * Dc + k0 + co, &Xs[buf][row][co]);
    }
  };
  // two-phase: issue ALL global loads, then convert+store (keeps 6 loads in
  // flight under a single loadcnt wait instead of one wait per load)
  auto stageW = [&](int buf, int k0) {
    int row = tid >> 2;
    int co  = (tid & 3) * 8;
    float4 f[6];
#pragma unroll
    for (int w = 0; w < 3; ++w) {
      const float* s = Wp[w] + (size_t)(n0 + row) * Dc + k0 + co;
      f[2 * w]     = *(const float4*)(s);
      f[2 * w + 1] = *(const float4*)(s + 4);
    }
#pragma unroll
    for (int w = 0; w < 3; ++w)
      *(v8bf*)&Ws[w][buf][row][co] = pack8(f[2 * w], f[2 * w + 1]);
  };
  auto fragA = [&](int buf) {
    const float* xr = &Xs[buf][mw * 16 + r][0];
    float4 a = *(const float4*)(xr + half * 8);
    float4 b = *(const float4*)(xr + half * 8 + 4);
    float4 c = *(const float4*)(xr + 16 + half * 8);
    float4 d = *(const float4*)(xr + 16 + half * 8 + 4);
    return cat16(pack8(a, b), pack8(c, d));
  };
  auto fragB = [&](int w, int buf, int ro) {
    const __bf16* wr = &Ws[w][buf][nw * 32 + ro + r][0];
    v8bf lo = *(const v8bf*)(wr + half * 8);
    v8bf hi = *(const v8bf*)(wr + 16 + half * 8);
    return cat16(lo, hi);
  };

  v8f aq0 = zero8(), aq1 = zero8();
  v8f ak0 = zero8(), ak1 = zero8();
  v8f av0 = zero8(), av1 = zero8();

  stageX(0, 0);
  stageW(0, 0);
  wait_async0();
  __syncthreads();

  for (int ks = 0; ks < Dc / 32; ++ks) {
    const int  buf = ks & 1;
    const bool nx  = (ks + 1) < Dc / 32;
    if (nx) stageX(buf ^ 1, (ks + 1) * 32);   // async copies overlap wmma

    v16bf a = fragA(buf);
    // per-weight fragment pairs -> back-to-back wmma bursts
    v16bf q0f = fragB(0, buf, 0), q1f = fragB(0, buf, 16);
    aq0 = wmma_bf16(a, q0f, aq0);
    aq1 = wmma_bf16(a, q1f, aq1);
    v16bf k0f = fragB(1, buf, 0), k1f = fragB(1, buf, 16);
    ak0 = wmma_bf16(a, k0f, ak0);
    ak1 = wmma_bf16(a, k1f, ak1);
    v16bf v0f = fragB(2, buf, 0), v1f = fragB(2, buf, 16);
    av0 = wmma_bf16(a, v0f, av0);
    av1 = wmma_bf16(a, v1f, av1);

    if (nx) {
      stageW(buf ^ 1, (ks + 1) * 32);
      wait_async0();
      __syncthreads();
    }
  }

  // ---- epilogue: bias + scatter to [B,H,S,Hd] (Q,K) and [B,H,Hd,S] (V) ----
  const int cn0 = n0 + nw * 32 + r;
  const int cn1 = cn0 + 16;
  const float bq0 = bq[cn0], bq1 = bq[cn1];
  const float bk0 = bk[cn0], bk1 = bk[cn1];
  const float bv0 = bv[cn0], bv1 = bv[cn1];
  const int h0 = cn0 >> 6, hd0 = cn0 & 63;
  const int h1 = cn1 >> 6, hd1 = cn1 & 63;

#pragma unroll
  for (int j = 0; j < 8; ++j) {
    const int m = m0 + mw * 16 + half * 8 + j;
    const int b = m >> 11;
    const int s = m & (Sc - 1);
    size_t qi0 = (((size_t)(b * Hc + h0) * Sc) + s) * Hd + hd0;
    size_t qi1 = (((size_t)(b * Hc + h1) * Sc) + s) * Hd + hd1;
    Qb[qi0] = (__bf16)(aq0[j] + bq0);
    Qb[qi1] = (__bf16)(aq1[j] + bq1);
    Kb[qi0] = (__bf16)(ak0[j] + bk0);
    Kb[qi1] = (__bf16)(ak1[j] + bk1);
    size_t vi0 = (((size_t)(b * Hc + h0) * Hd) + hd0) * Sc + s;
    size_t vi1 = (((size_t)(b * Hc + h1) * Hd) + hd1) * Sc + s;
    Vb[vi0] = (__bf16)(av0[j] + bv0);
    Vb[vi1] = (__bf16)(av1[j] + bv1);
  }
}

// ---------------------------------------------------------------------------
// Kernel 2: flash attention.  Grid = (S/128, B*H).  Each wave owns 16 query
// rows, streams keys in steps of 32, online softmax state per row.
// ---------------------------------------------------------------------------
__global__ __launch_bounds__(256) void attn_kernel(
    const __bf16* __restrict__ Qb, const __bf16* __restrict__ Kb,
    const __bf16* __restrict__ Vb, const int* __restrict__ mask,
    __bf16* __restrict__ Ctx) {
  __shared__ __align__(16) __bf16 Ps[8][16][40];   // per-wave P re-layout (bf16)

  const int tid  = threadIdx.x;
  const int wave = tid >> 5;
  const int lane = tid & 31;
  const int r    = lane & 15;
  const int half = lane >> 4;

  const int bh = blockIdx.y;                // b*H + h
  const int b  = bh >> 4;
  const int h  = bh & 15;
  const int q0 = blockIdx.x * 128 + wave * 16;

  const __bf16* qp = Qb + ((size_t)bh * Sc + (q0 + r)) * Hd + half * 8;
  v16bf qf0 = frag_from_bf16(qp);
  v16bf qf1 = frag_from_bf16(qp + 32);

  float mrow[8], lrow[8];
  v8f o0 = zero8(), o1 = zero8(), o2 = zero8(), o3 = zero8();
#pragma unroll
  for (int j = 0; j < 8; ++j) { mrow[j] = -3.0e38f; lrow[j] = 0.f; }

  const __bf16* kbase = Kb + (size_t)bh * Sc * Hd;
  const __bf16* vbase = Vb + (size_t)bh * Hd * Sc;
  const int*    mbase = mask + (size_t)b * Sc;

  for (int kk0 = 0; kk0 < Sc; kk0 += 32) {
    if (kk0 + 32 < Sc) {                     // prefetch next K/V tile rows
      __builtin_prefetch(kbase + (size_t)(kk0 + 32 + r) * Hd, 0, 1);
      __builtin_prefetch(vbase + (size_t)r * Sc + kk0 + 32, 0, 1);
    }
    // ---- scores: S[16 q x 32 k] = Q (16x64) * K^T ----
    const __bf16* kp0 = kbase + (size_t)(kk0 + r)      * Hd + half * 8;
    const __bf16* kp1 = kbase + (size_t)(kk0 + 16 + r) * Hd + half * 8;
    v16bf kf00 = frag_from_bf16(kp0);
    v16bf kf01 = frag_from_bf16(kp0 + 32);
    v16bf kf10 = frag_from_bf16(kp1);
    v16bf kf11 = frag_from_bf16(kp1 + 32);

    v8f s0 = zero8(), s1 = zero8();
    s0 = wmma_bf16(qf0, kf00, s0);
    s0 = wmma_bf16(qf1, kf01, s0);
    s1 = wmma_bf16(qf0, kf10, s1);
    s1 = wmma_bf16(qf1, kf11, s1);

    const float madd0 = mbase[kk0 + r]      ? 0.f : -1.0e9f;
    const float madd1 = mbase[kk0 + 16 + r] ? 0.f : -1.0e9f;
    s0 = s0 * 0.125f + madd0;                // 1/sqrt(Hd)
    s1 = s1 * 0.125f + madd1;

    // ---- online softmax; write P (bf16) into per-wave LDS [row][col] ----
#pragma unroll
    for (int j = 0; j < 8; ++j) {
      float rmax = rmax16(fmaxf(s0[j], s1[j]));
      float mold = mrow[j];
      float mnew = fmaxf(mold, rmax);
      float alpha = __expf(mold - mnew);
      float p0 = __expf(s0[j] - mnew);
      float p1 = __expf(s1[j] - mnew);
      lrow[j] = lrow[j] * alpha + rsum16(p0 + p1);
      mrow[j] = mnew;
      o0[j] *= alpha; o1[j] *= alpha; o2[j] *= alpha; o3[j] *= alpha;
      Ps[wave][j + half * 8][r]      = (__bf16)p0;
      Ps[wave][j + half * 8][16 + r] = (__bf16)p1;
    }
    asm volatile("s_wait_dscnt 0x0" ::: "memory");

    // ---- re-read P as an A-fragment (16 x 32), already bf16 ----
    const __bf16* rowp = &Ps[wave][r][0];
    v16bf pf = cat16(*(const v8bf*)(rowp + half * 8),
                     *(const v8bf*)(rowp + 16 + half * 8));

    // ---- O += P @ V   (V^T rows contiguous along kk) ----
    const __bf16* vp = vbase + (size_t)r * Sc + kk0 + half * 8;
    v16bf vf0 = frag_from_bf16(vp);
    v16bf vf1 = frag_from_bf16(vp + 16 * Sc);
    v16bf vf2 = frag_from_bf16(vp + 32 * Sc);
    v16bf vf3 = frag_from_bf16(vp + 48 * Sc);
    o0 = wmma_bf16(pf, vf0, o0);
    o1 = wmma_bf16(pf, vf1, o1);
    o2 = wmma_bf16(pf, vf2, o2);
    o3 = wmma_bf16(pf, vf3, o3);
  }

  // ---- normalize and store context bf16 [B,S,D] ----
#pragma unroll
  for (int j = 0; j < 8; ++j) {
    const float inv = 1.0f / lrow[j];
    const int q = q0 + half * 8 + j;
    size_t base = ((size_t)b * Sc + q) * Dc + h * Hd;
    Ctx[base +      r] = (__bf16)(o0[j] * inv);
    Ctx[base + 16 + r] = (__bf16)(o1[j] * inv);
    Ctx[base + 32 + r] = (__bf16)(o2[j] * inv);
    Ctx[base + 48 + r] = (__bf16)(o3[j] * inv);
  }
}

// ---------------------------------------------------------------------------
// Kernel 3: out projection + residual.  y = x + ctx @ Wo.T + bo  (fp32 y).
// Ctx tile async-staged (already bf16); Wo staged as bf16 once per K-step.
// ---------------------------------------------------------------------------
__global__ __launch_bounds__(256) void oproj_kernel(
    const __bf16* __restrict__ Ctx, const float* __restrict__ Wo,
    const float* __restrict__ bo, const float* __restrict__ x,
    float* __restrict__ Y) {
  __shared__ __align__(16) __bf16 Cs[2][64][40];
  __shared__ __align__(16) __bf16 Os[2][64][40];

  const int tid  = threadIdx.x;
  const int wave = tid >> 5;
  const int lane = tid & 31;
  const int r    = lane & 15;
  const int half = lane >> 4;
  const int mw   = wave & 3;
  const int nw   = wave >> 2;
  const int m0   = blockIdx.y * 64;
  const int n0   = blockIdx.x * 64;

  auto stageC = [&](int buf, int k0) {
    int row = tid >> 2;
    int co  = (tid & 3) * 8;                 // 8 bf16 = 16B per thread
    async_b128(Ctx + (size_t)(m0 + row) * Dc + k0 + co, &Cs[buf][row][co]);
  };
  auto stageW = [&](int buf, int k0) {
    int row = tid >> 2;
    int co  = (tid & 3) * 8;
    const float* s = Wo + (size_t)(n0 + row) * Dc + k0 + co;
    float4 f0 = *(const float4*)(s);
    float4 f1 = *(const float4*)(s + 4);
    *(v8bf*)&Os[buf][row][co] = pack8(f0, f1);
  };

  v8f a0 = zero8(), a1 = zero8();

  stageC(0, 0);
  stageW(0, 0);
  wait_async0();
  __syncthreads();

  for (int ks = 0; ks < Dc / 32; ++ks) {
    const int  buf = ks & 1;
    const bool nx  = (ks + 1) < Dc / 32;
    if (nx) stageC(buf ^ 1, (ks + 1) * 32);

    const __bf16* cr = &Cs[buf][mw * 16 + r][0];
    v16bf a = cat16(*(const v8bf*)(cr + half * 8),
                    *(const v8bf*)(cr + 16 + half * 8));
    const __bf16* w0 = &Os[buf][nw * 32 + r][0];
    const __bf16* w1 = &Os[buf][nw * 32 + 16 + r][0];
    v16bf b0 = cat16(*(const v8bf*)(w0 + half * 8),
                     *(const v8bf*)(w0 + 16 + half * 8));
    v16bf b1 = cat16(*(const v8bf*)(w1 + half * 8),
                     *(const v8bf*)(w1 + 16 + half * 8));
    a0 = wmma_bf16(a, b0, a0);
    a1 = wmma_bf16(a, b1, a1);

    if (nx) {
      stageW(buf ^ 1, (ks + 1) * 32);
      wait_async0();
      __syncthreads();
    }
  }

  const int cn0 = n0 + nw * 32 + r;
  const int cn1 = cn0 + 16;
  const float bo0 = bo[cn0], bo1 = bo[cn1];
#pragma unroll
  for (int j = 0; j < 8; ++j) {
    const int m = m0 + mw * 16 + half * 8 + j;
    Y[(size_t)m * Dc + cn0] = a0[j] + bo0 + x[(size_t)m * Dc + cn0];
    Y[(size_t)m * Dc + cn1] = a1[j] + bo1 + x[(size_t)m * Dc + cn1];
  }
}

// ---------------------------------------------------------------------------
// Kernel 4: LayerNorm over D=1024 per row.  One block (256 thr) per row.
// ---------------------------------------------------------------------------
__global__ __launch_bounds__(256) void ln_kernel(
    const float* __restrict__ Y, const float* __restrict__ gamma,
    const float* __restrict__ beta, float* __restrict__ out) {
  __shared__ float redS[8], redQ[8];
  const int row  = blockIdx.x;
  const int tid  = threadIdx.x;
  const int wave = tid >> 5;
  const int lane = tid & 31;

  const float4 v = ((const float4*)(Y + (size_t)row * Dc))[tid];
  float s = v.x + v.y + v.z + v.w;
  float q = v.x * v.x + v.y * v.y + v.z * v.z + v.w * v.w;
#pragma unroll
  for (int m = 16; m >= 1; m >>= 1) {
    s += __shfl_xor(s, m);
    q += __shfl_xor(q, m);
  }
  if (lane == 0) { redS[wave] = s; redQ[wave] = q; }
  __syncthreads();
  float S = 0.f, Q = 0.f;
#pragma unroll
  for (int i = 0; i < 8; ++i) { S += redS[i]; Q += redQ[i]; }
  const float mu   = S * (1.0f / Dc);
  const float var  = Q * (1.0f / Dc) - mu * mu;
  const float rstd = rsqrtf(var + 1e-5f);

  const float4 g  = ((const float4*)gamma)[tid];
  const float4 be = ((const float4*)beta)[tid];
  float4 o;
  o.x = (v.x - mu) * rstd * g.x + be.x;
  o.y = (v.y - mu) * rstd * g.y + be.y;
  o.z = (v.z - mu) * rstd * g.z + be.z;
  o.w = (v.w - mu) * rstd * g.w + be.w;
  ((float4*)(out + (size_t)row * Dc))[tid] = o;
}

// ---------------------------------------------------------------------------
// Launcher
// ---------------------------------------------------------------------------
extern "C" void kernel_launch(void* const* d_in, const int* in_sizes, int n_in,
                              void* d_out, int out_size, void* d_ws, size_t ws_size,
                              hipStream_t stream) {
  const float* x     = (const float*)d_in[0];
  const int*   mask  = (const int*)  d_in[1];
  const float* Wq    = (const float*)d_in[2];
  const float* bq    = (const float*)d_in[3];
  const float* Wk    = (const float*)d_in[4];
  const float* bk    = (const float*)d_in[5];
  const float* Wv    = (const float*)d_in[6];
  const float* bv    = (const float*)d_in[7];
  const float* Wo    = (const float*)d_in[8];
  const float* bo    = (const float*)d_in[9];
  const float* gamma = (const float*)d_in[10];
  const float* beta  = (const float*)d_in[11];

  char* ws = (char*)d_ws;
  const size_t ELEMS = (size_t)Mtot * Dc;          // 8 Mi elements
  __bf16* Qb  = (__bf16*)(ws);                     // 16 MB
  __bf16* Kb  = (__bf16*)(ws + 2 * ELEMS);         // 16 MB
  __bf16* Vb  = (__bf16*)(ws + 4 * ELEMS);         // 16 MB
  __bf16* Ctx = (__bf16*)(ws + 6 * ELEMS);         // 16 MB
  float*  Y   = (float*) (ws + 8 * ELEMS);         // 32 MB

  dim3 blk(256);
  dim3 g1(Dc / 64, Mtot / 64);                     // (16, 128)
  qkv_kernel<<<g1, blk, 0, stream>>>(x, Wq, bq, Wk, bk, Wv, bv, Qb, Kb, Vb);

  dim3 g2(Sc / 128, Bc * Hc);                      // (16, 64)
  attn_kernel<<<g2, blk, 0, stream>>>(Qb, Kb, Vb, mask, Ctx);

  oproj_kernel<<<g1, blk, 0, stream>>>(Ctx, Wo, bo, x, Y);

  ln_kernel<<<Mtot, blk, 0, stream>>>(Y, gamma, beta, (float*)d_out);
}
